// AttentionPolicyHead_2216203125394
// MI455X (gfx1250) — matrix operand order repaired and checked by reading, not compiled, not run on hardware
//
#include <hip/hip_runtime.h>
#include <hip/hip_bf16.h>

typedef __attribute__((ext_vector_type(16))) __bf16 v16bf;
typedef __attribute__((ext_vector_type(8)))  float  v8f;

#define EMBED   1024
#define PDIM    128
#define NCHUNK  32                       // EMBED / 32
#define NEG_INF_F (-1.0e9f)
#define QK_SCALE 0.08838834764831845f    // 128^-0.5

// LDS strides (elements), padded to spread banks
#define XS_STRIDE  40    // 32 + 8 pad  (bf16)
#define WT_STRIDE  40    // 32 + 8 pad  (bf16)
#define QK_STRIDE  136   // 128 + 8 pad (bf16)
#define LFT_STRIDE 65    // 64 + 1 pad  (f32)

// phase-1 (double buffered): buffer b at byte offset b*25600; WT at +5120 inside buffer
#define P1_BUF_BYTES 25600
#define WT_IN_BUF    5120                // phase1 end 51200
// phase-2 union
#define QS_OFF   0                       // [64][136] bf16 -> 17408
#define KS_OFF   17408                   // [64][136] bf16 -> 17408
#define LFT_OFF  34816                   // [64][65]  f32  -> 16640
#define UPL_OFF  51456                   // [8][9]    f32  -> 288
#define SMEM_BYTES 51744

#define WT_ROWS  256                     // Wq n=0..127, Wk n=128..255

__device__ __constant__ int c_dr[8]  = { 1, 1, 0,-1,-1,-1, 0, 1};
__device__ __constant__ int c_dc[8]  = { 0, 1, 1, 1, 0,-1,-1,-1};
__device__ __constant__ int c_kdr[8] = { 2, 1,-1,-2,-2,-1, 1, 2};
__device__ __constant__ int c_kdc[8] = { 1, 2, 2, 1,-1,-2,-2,-1};

union FragU { uint4 u[2]; v16bf v; };

// one v_cvt_pk_bf16_f32 per pair (dst.lo = cvt(a), dst.hi = cvt(b))
__device__ inline unsigned pack_bf16x2(float a, float b) {
    unsigned r;
    asm("v_cvt_pk_bf16_f32 %0, %1, %2" : "=v"(r) : "v"(a), "v"(b));
    return r;
}

__device__ inline FragU load_frag(const __bf16* p0) {
    FragU f;
    f.u[0] = *(const uint4*)p0;
    f.u[1] = *(const uint4*)(p0 + 8);
    return f;
}

// One-time prep: Wt[n][d] = bf16( (n<128 ? Wq : Wk)[d][n&127] )   (n-major, K-contiguous)
__global__ __launch_bounds__(256)
void weights_prep_kernel(const float* __restrict__ Wq, const float* __restrict__ Wk,
                         __bf16* __restrict__ wt) {
    const int n = blockIdx.x;                           // 0..255
    const float* src = (n < 128) ? (Wq + n) : (Wk + (n - 128));
    for (int d = threadIdx.x; d < EMBED; d += 256)
        wt[(size_t)n * EMBED + d] = (__bf16)src[(size_t)d * PDIM];
}

// Async global->LDS 16B copy (CDNA5 ASYNCcnt path; no VGPR round trip)
__device__ inline void async_copy16(const __bf16* gsrc, __bf16* ldst) {
    const unsigned lo = (unsigned)(unsigned long long)ldst;  // low 32b = LDS offset
    asm volatile("global_load_async_to_lds_b128 %0, %1, off"
                 :: "v"(lo), "v"(gsrc) : "memory");
}

// Stage one 32-wide K-chunk of Wt (256 rows x 64B) into padded LDS, 4 async b128/thread
__device__ inline void stage_wt_async(const __bf16* __restrict__ wt_g,
                                      __bf16* wt_lds, int d0, int t) {
    const int quad = t & 3;     // 16B piece within row chunk
    const int n0   = t >> 2;    // 0..63
    #pragma unroll
    for (int r = 0; r < 4; ++r) {
        const int n = n0 + r * 64;
        async_copy16(wt_g + (size_t)n * EMBED + d0 + quad * 8,
                     wt_lds + n * WT_STRIDE + quad * 8);
    }
}

__global__ __launch_bounds__(256)
void policy_head_kernel(const float* __restrict__ x,
                        const __bf16* __restrict__ wt_g,
                        const float* __restrict__ bq, const float* __restrict__ bk,
                        const float* __restrict__ Wu, const float* __restrict__ bu,
                        float* __restrict__ out)
{
    __shared__ __align__(16) unsigned char smem[SMEM_BYTES];
    __bf16* qs  = (__bf16*)(smem + QS_OFF);
    __bf16* ks  = (__bf16*)(smem + KS_OFF);
    float*  lft = (float*) (smem + LFT_OFF);
    float*  upl = (float*) (smem + UPL_OFF);

    const int t    = threadIdx.x;
    const int lane = t & 31;
    const int w    = t >> 5;
    const int b    = blockIdx.x;
    const int half = lane >> 4;
    const int l15  = lane & 15;

    const int strip = w >> 1;        // 16-row strip
    const int sel   = w & 1;         // 0 = q, 1 = k
    const int m0    = strip * 16;

    const float* xb = x + (size_t)b * 64 * EMBED;

    // x staging role (fixed per thread): 2 rows x one float4 (8 threads per row)
    const int qd   = t & 7;          // float4 index within 32-wide row chunk
    const int rowh = t >> 3;         // 0..31; rows rowh and rowh+32

    v8f zero;
    #pragma unroll
    for (int i = 0; i < 8; ++i) zero[i] = 0.0f;
    v8f acc[8];
    #pragma unroll
    for (int j = 0; j < 8; ++j) acc[j] = zero;

    // -------- prologue: stage chunk 0 into buffer 0 --------
    {
        __bf16* xs0 = (__bf16*)smem;
        __bf16* wt0 = (__bf16*)(smem + WT_IN_BUF);
        stage_wt_async(wt_g, wt0, 0, t);
        #pragma unroll
        for (int r = 0; r < 2; ++r) {
            const int row = rowh + r * 32;
            const float4 v = *(const float4*)(xb + (size_t)row * EMBED + qd * 4);
            uint2 pk;
            pk.x = pack_bf16x2(v.x, v.y);
            pk.y = pack_bf16x2(v.z, v.w);
            *(uint2*)(xs0 + row * XS_STRIDE + qd * 4) = pk;
        }
    }
    asm volatile("s_wait_asynccnt 0x0" ::: "memory");
    __syncthreads();

    // -------- Phase 1: q/k projections, double-buffered K-loop --------
    for (int c = 0; c < NCHUNK; ++c) {
        const int cur = c & 1, nxt = cur ^ 1;
        unsigned char* base_cur = smem + cur * P1_BUF_BYTES;
        unsigned char* base_nxt = smem + nxt * P1_BUF_BYTES;
        __bf16* xs_c = (__bf16*)base_cur;
        __bf16* wt_c = (__bf16*)(base_cur + WT_IN_BUF);
        __bf16* xs_n = (__bf16*)base_nxt;
        __bf16* wt_n = (__bf16*)(base_nxt + WT_IN_BUF);

        const bool have_next = (c + 1 < NCHUNK);
        float4 pre[2];
        if (have_next) {
            const int d1 = (c + 1) * 32;
            stage_wt_async(wt_g, wt_n, d1, t);              // async -> LDS
            #pragma unroll
            for (int r = 0; r < 2; ++r) {                   // global -> regs
                const int row = rowh + r * 32;
                pre[r] = *(const float4*)(xb + (size_t)row * EMBED + d1 + qd * 4);
            }
        }

        // A fragment (16x32 bf16)
        FragU a = load_frag(xs_c + (m0 + l15) * XS_STRIDE + half * 16);

        // depth-2 software-pipelined B fragments -> 8 WMMAs
        const __bf16* wbase = wt_c + (sel * 128 + l15) * WT_STRIDE + half * 16;
        FragU b0f = load_frag(wbase);
        FragU b1f = load_frag(wbase + 16 * WT_STRIDE);
        #pragma unroll
        for (int j = 0; j < 8; ++j) {
            FragU bnext;
            if (j < 6) bnext = load_frag(wbase + (j + 2) * 16 * WT_STRIDE);
            acc[j] = __builtin_amdgcn_wmma_f32_16x16x32_bf16(
                         false, a.v, false, b0f.v, (short)0, acc[j], false, false);
            b0f = b1f;
            if (j < 6) b1f = bnext;
        }

        if (have_next) {
            #pragma unroll
            for (int r = 0; r < 2; ++r) {                   // regs -> LDS (bf16)
                const int row = rowh + r * 32;
                uint2 pk;
                pk.x = pack_bf16x2(pre[r].x, pre[r].y);
                pk.y = pack_bf16x2(pre[r].z, pre[r].w);
                *(uint2*)(xs_n + row * XS_STRIDE + qd * 4) = pk;
            }
            asm volatile("s_wait_asynccnt 0x0" ::: "memory");
        }
        __syncthreads();
    }

    // -------- bias + bf16 convert + store q/k row-major [64][128] --------
    {
        const float* bias = sel ? bk : bq;
        __bf16* dst = sel ? ks : qs;
        #pragma unroll
        for (int j = 0; j < 8; ++j) {
            const float bv = bias[j * 16 + l15];
            #pragma unroll
            for (int r = 0; r < 8; ++r) {
                const int m = m0 + half * 8 + r;
                dst[m * QK_STRIDE + j * 16 + l15] = (__bf16)(acc[j][r] + bv);
            }
        }
    }
    __syncthreads();

    // -------- Phase 2: logits = q @ k^T * SCALE (64x64, K=128) --------
    {
        const int mm0 = (w >> 1) * 16;
        const int nn0 = (w & 1) * 32;
        v8f c2a = zero, c2b = zero;
        #pragma unroll
        for (int kc = 0; kc < 4; ++kc) {
            FragU a2 = load_frag(qs + (mm0 + l15) * QK_STRIDE + kc * 32 + half * 16);
            FragU b0 = load_frag(ks + (nn0 + l15) * QK_STRIDE + kc * 32 + half * 16);
            FragU b1 = load_frag(ks + (nn0 + 16 + l15) * QK_STRIDE + kc * 32 + half * 16);
            c2a = __builtin_amdgcn_wmma_f32_16x16x32_bf16(
                      false, a2.v, false, b0.v, (short)0, c2a, false, false);
            c2b = __builtin_amdgcn_wmma_f32_16x16x32_bf16(
                      false, a2.v, false, b1.v, (short)0, c2b, false, false);
        }
        #pragma unroll
        for (int r = 0; r < 8; ++r) {
            const int m = mm0 + half * 8 + r;
            lft[m * LFT_STRIDE + nn0 + l15]      = c2a[r] * QK_SCALE;
            lft[m * LFT_STRIDE + nn0 + 16 + l15] = c2b[r] * QK_SCALE;
        }
    }

    // -------- Promotion logits: rows 48..55 only, one wave per row --------
    {
        const int s = 48 + w;
        const float* xr = xb + (size_t)s * EMBED;
        float ua[9];
        #pragma unroll
        for (int n = 0; n < 9; ++n) ua[n] = 0.0f;
        for (int k = lane; k < EMBED; k += 32) {
            const float xv = xr[k];
            #pragma unroll
            for (int n = 0; n < 9; ++n) ua[n] += xv * Wu[(size_t)k * 9 + n];
        }
        #pragma unroll
        for (int n = 0; n < 9; ++n) {
            float v = ua[n];
            #pragma unroll
            for (int off = 16; off > 0; off >>= 1) v += __shfl_xor(v, off, 32);
            if (lane == 0) upl[w * 9 + n] = v + bu[n];
        }
    }
    __syncthreads();

    // -------- Epilogue: gather into 73-plane layout + mask --------
    {
        float* ob = out + (size_t)b * (64 * 73);
        for (int idx = t; idx < 64 * 73; idx += 256) {
            const int f = idx / 73;
            const int p = idx - f * 73;
            float val;
            if (p < 64) {
                const int r = f >> 3, c = f & 7;
                int nr, nc;
                if (p < 56) {
                    const int d    = p / 7;
                    const int dist = p - d * 7 + 1;
                    nr = r + c_dr[d] * dist;
                    nc = c + c_dc[d] * dist;
                } else {
                    const int i = p - 56;
                    nr = r + c_kdr[i];
                    nc = c + c_kdc[i];
                }
                val = (nr >= 0 && nr < 8 && nc >= 0 && nc < 8)
                        ? lft[f * LFT_STRIDE + nr * 8 + nc] : NEG_INF_F;
            } else {
                const int n = p - 64;
                val = (f >= 48 && f < 56) ? upl[(f - 48) * 9 + n] : NEG_INF_F;
            }
            ob[idx] = val;
        }
    }
}

extern "C" void kernel_launch(void* const* d_in, const int* in_sizes, int n_in,
                              void* d_out, int out_size, void* d_ws, size_t ws_size,
                              hipStream_t stream) {
    const float* x  = (const float*)d_in[0];
    const float* Wq = (const float*)d_in[1];
    const float* bq = (const float*)d_in[2];
    const float* Wk = (const float*)d_in[3];
    const float* bk = (const float*)d_in[4];
    const float* Wu = (const float*)d_in[5];
    const float* bu = (const float*)d_in[6];
    float* out = (float*)d_out;
    __bf16* wt = (__bf16*)d_ws;                     // 512 KB bf16 transposed weights

    weights_prep_kernel<<<WT_ROWS, 256, 0, stream>>>(Wq, Wk, wt);

    const int B = in_sizes[0] / (64 * EMBED);       // 1024
    policy_head_kernel<<<B, 256, 0, stream>>>(x, wt, bq, bk, Wu, bu, out);
}
